// SamVisionAttention_51342039056568
// MI455X (gfx1250) — matrix operand order repaired and checked by reading, not compiled, not run on hardware
//
#include <hip/hip_runtime.h>
#include <hip/hip_bf16.h>
#include <stdint.h>

// ---------------------------------------------------------------------------
// SAM vision attention for MI455X (gfx1250, wave32, WMMA, async global->LDS).
// Pipeline:
//   1) convert x / w_qkv / w_proj f32 -> bf16
//   2) QKV GEMM (bf16 WMMA, f32 acc, async A-tile staging) -> q(scaled)/k/v
//   3) rel_h / rel_w bias tiles (VALU, tiny FLOPs)
//   4) fused flash-attention (bf16 WMMA QK^T and PV, f32 online softmax,
//      async K-tile staging + prefetch of next tiles, decomposed rel-pos
//      bias added to S tiles) -> attn_out bf16
//   5) output projection GEMM (bf16 WMMA) + bias -> f32 d_out
// ---------------------------------------------------------------------------

typedef __bf16 bf16;
typedef __attribute__((ext_vector_type(16))) bf16  v16bf;
typedef __attribute__((ext_vector_type(8)))  bf16  bf16x8;
typedef __attribute__((ext_vector_type(8)))  float v8f;

#define DIM       768
#define NHEADS    12
#define HD        64
#define N_TOK     4096   // 64*64

__device__ __forceinline__ bf16 f2bf(float f) {
  uint32_t u = __builtin_bit_cast(uint32_t, f);
  uint32_t r = u + 0x7FFFu + ((u >> 16) & 1u);   // round-to-nearest-even
  uint16_t h = (uint16_t)(r >> 16);
  return __builtin_bit_cast(bf16, h);
}
__device__ __forceinline__ float bf2f(bf16 b) {
  uint16_t h = __builtin_bit_cast(uint16_t, b);
  uint32_t u = ((uint32_t)h) << 16;
  return __builtin_bit_cast(float, u);
}

__device__ __forceinline__ v8f wmma_bf16(v16bf a, v16bf b, v8f c) {
  // D = A(16x32 bf16) * B(32x16 bf16) + C(16x16 f32)
  return __builtin_amdgcn_wmma_f32_16x16x32_bf16(false, a, false, b,
                                                 (short)0, c, false, false);
}

// ---- CDNA5 async global->LDS copy (ASYNCcnt-tracked, bypasses VGPRs) ------
typedef __attribute__((address_space(3))) void lds_void;
__device__ __forceinline__ uint32_t lds_off(const void* p) {
  return (uint32_t)(uintptr_t)(lds_void*)(p);
}
__device__ __forceinline__ void async_b128(uint32_t dst_lds, const void* src) {
  uint64_t ga = (uint64_t)(uintptr_t)src;
  asm volatile("global_load_async_to_lds_b128 %0, %1, off"
               :: "v"(dst_lds), "v"(ga) : "memory");
}
__device__ __forceinline__ void wait_async() {
  asm volatile("s_wait_asynccnt 0x0" ::: "memory");
}

// A fragment: LDS row-major [M][K], ISA 16-bit A layout:
// lanes 0-15: elem e<8 -> K=k0+e, e>=8 -> K=k0+16+(e-8); lanes 16-31: +8.
__device__ __forceinline__ v16bf frag_a(const bf16* base, int row0, int stride, int k0) {
  const int lane = threadIdx.x & 31;
  const bf16* p = base + (row0 + (lane & 15)) * stride + k0 + ((lane >> 4) << 3);
  v16bf a;
#pragma unroll
  for (int e = 0; e < 8; e++) { a[e] = p[e]; a[e + 8] = p[e + 16]; }
  return a;
}

// B fragment from LDS holding B TRANSPOSED as [N][K]:
// lanes 0-15: elem e<8 -> K=k0+e, e>=8 -> K=k0+16+(e-8); lanes 16-31: +8; col N = lane&15.
__device__ __forceinline__ v16bf frag_b(const bf16* baseT, int col0, int stride, int k0) {
  const int lane = threadIdx.x & 31;
  const bf16* p = baseT + (col0 + (lane & 15)) * stride + k0 + ((lane >> 4) << 3);
  v16bf b;
#pragma unroll
  for (int e = 0; e < 8; e++) { b[e] = p[e]; b[e + 8] = p[e + 16]; }
  return b;
}

// --------------------------- f32 -> bf16 convert ---------------------------
__global__ __launch_bounds__(256) void cvt_kernel(const float* __restrict__ in,
                                                  bf16* __restrict__ out, int n) {
  int i = blockIdx.x * 256 + threadIdx.x;
  if (i < n) out[i] = f2bf(in[i]);
}

// ------------------------------- QKV GEMM ----------------------------------
// C(4096 x 2304) = x_bf(4096x768) @ wqkv_bf(768x2304) + b_qkv
// block tile 128x64, K-step 32, 8 waves in 4x2 grid (each wave 32x32).
__global__ __launch_bounds__(256) void qkv_gemm_kernel(
    const bf16* __restrict__ A, const bf16* __restrict__ B,
    const float* __restrict__ bias,
    bf16* __restrict__ qb, bf16* __restrict__ kb, bf16* __restrict__ vb) {
  __shared__ bf16 a_s[128 * 40];   // [M][K] stride 40
  __shared__ bf16 b_s[64 * 40];    // [N][K] (transposed) stride 40
  const int tid  = threadIdx.x;
  const int lane = tid & 31, wid = tid >> 5;
  const int wm = wid >> 1, wn = wid & 1;
  const int m0 = blockIdx.y * 128, n0 = blockIdx.x * 64;

  v8f acc[2][2] = {};
  for (int k0 = 0; k0 < DIM; k0 += 32) {
    for (int i = tid; i < 512; i += 256) {            // A: 128x32 bf16, async
      int r = i >> 2, kc = (i & 3) << 3;
      async_b128(lds_off(&a_s[r * 40 + kc]),
                 &A[(m0 + r) * DIM + k0 + kc]);
    }
    {                                                 // B: 32x64, store transposed
      int kr = tid >> 3, nc = (tid & 7) << 3;
      bf16x8 v = *(const bf16x8*)&B[(k0 + kr) * 2304 + n0 + nc];
#pragma unroll
      for (int j = 0; j < 8; j++) b_s[(nc + j) * 40 + kr] = v[j];
    }
    wait_async();
    __syncthreads();
    v16bf af[2], bf_[2];
    af[0] = frag_a(a_s, wm * 32,      40, 0);
    af[1] = frag_a(a_s, wm * 32 + 16, 40, 0);
    bf_[0] = frag_b(b_s, wn * 32,      40, 0);
    bf_[1] = frag_b(b_s, wn * 32 + 16, 40, 0);
#pragma unroll
    for (int tm = 0; tm < 2; tm++)
#pragma unroll
      for (int tn = 0; tn < 2; tn++)
        acc[tm][tn] = wmma_bf16(af[tm], bf_[tn], acc[tm][tn]);
    __syncthreads();
  }

  // epilogue: bias, split into q/k/v per-head [head][n][c], scale q by 0.125
  const int roff = (lane >> 4) << 3;
  const int ncol = lane & 15;
#pragma unroll
  for (int tm = 0; tm < 2; tm++)
#pragma unroll
    for (int tn = 0; tn < 2; tn++) {
      int gn = n0 + wn * 32 + tn * 16 + ncol;         // 0..2303
      float bv    = bias[gn];
      int which   = gn / DIM;
      int rem     = gn - which * DIM;
      int head    = rem >> 6;
      int c       = rem & 63;
      bf16* dst   = (which == 0) ? qb : (which == 1) ? kb : vb;
      float scale = (which == 0) ? 0.125f : 1.0f;     // HD^-0.5
#pragma unroll
      for (int r = 0; r < 8; r++) {
        int m = m0 + wm * 32 + tm * 16 + r + roff;
        float v = (acc[tm][tn][r] + bv) * scale;
        dst[(head * N_TOK + m) * HD + c] = f2bf(v);
      }
    }
}

// ------------------------- rel_pos bias kernels ----------------------------
// rel_h[head][h][w][kh] = 8 * sum_c q_bf[head][h*64+w][c] * rel_pos_h[h-kh+63][c]
__global__ __launch_bounds__(256) void relh_kernel(const bf16* __restrict__ qg,
                                                   const float* __restrict__ rph,
                                                   float* __restrict__ out) {
  __shared__ bf16  q_s[64 * 64];
  __shared__ float r_s[64 * 64];
  const int head = blockIdx.x >> 6, h = blockIdx.x & 63;
  const int tid = threadIdx.x;
  const bf16* qp = qg + (head * N_TOK + h * 64) * HD;
  for (int i = tid; i < 512; i += 256) {
    int w = i >> 3, cc = (i & 7) << 3;
    *(bf16x8*)&q_s[w * 64 + cc] = *(const bf16x8*)&qp[w * 64 + cc];
  }
  const float* rp = rph + h * 64;            // rows h..h+63, contiguous
  for (int i = tid; i < 1024; i += 256)
    *(float4*)&r_s[i * 4] = *(const float4*)&rp[i * 4];
  __syncthreads();
  for (int i = tid; i < 4096; i += 256) {
    int w = i >> 6, k = i & 63;
    const bf16*  qr = &q_s[w * 64];
    const float* rr = &r_s[(63 - k) * 64];   // row index h-k+63 -> local 63-k
    float s = 0.f;
#pragma unroll
    for (int c = 0; c < 64; c++) s += bf2f(qr[c]) * rr[c];
    out[(head * 64 + h) * 4096 + w * 64 + k] = 8.0f * s;  // undo q scale
  }
}

// rel_w[head][h][w][kw] = 8 * sum_c q_bf[head][h*64+w][c] * rel_pos_w[w-kw+63][c]
__global__ __launch_bounds__(256) void relw_kernel(const bf16* __restrict__ qg,
                                                   const float* __restrict__ rpw,
                                                   bf16* __restrict__ out) {
  __shared__ bf16  q_s[64 * 64];             // [h][c]
  __shared__ float r_s[64 * 64];
  const int head = blockIdx.x >> 6, w = blockIdx.x & 63;
  const int tid = threadIdx.x;
  for (int i = tid; i < 512; i += 256) {
    int hr = i >> 3, cc = (i & 7) << 3;
    *(bf16x8*)&q_s[hr * 64 + cc] =
        *(const bf16x8*)&qg[(head * N_TOK + hr * 64 + w) * HD + cc];
  }
  const float* rp = rpw + w * 64;
  for (int i = tid; i < 1024; i += 256)
    *(float4*)&r_s[i * 4] = *(const float4*)&rp[i * 4];
  __syncthreads();
  for (int i = tid; i < 4096; i += 256) {
    int hr = i >> 6, k = i & 63;
    const bf16*  qr = &q_s[hr * 64];
    const float* rr = &r_s[(63 - k) * 64];
    float s = 0.f;
#pragma unroll
    for (int c = 0; c < 64; c++) s += bf2f(qr[c]) * rr[c];
    out[(head * 64 + hr) * 4096 + w * 64 + k] = f2bf(8.0f * s);
  }
}

// --------------------------- flash attention -------------------------------
// One WG per (head, q-block of 64 rows == one image row h=qb). 8 waves,
// wave (wm,wn): rows 16*wm, cols 32*wn. 64 iterations over 64-key blocks.
__global__ __launch_bounds__(256) void flash_kernel(
    const bf16* __restrict__ qg, const bf16* __restrict__ kg,
    const bf16* __restrict__ vg, const float* __restrict__ relh_g,
    const bf16* __restrict__ relw_g, bf16* __restrict__ outg) {
  __shared__ bf16  q_s[64 * 64];     // [w][c]
  __shared__ bf16  kt_s[64 * 64];    // [kw][c] == B^T for S = q k^T
  __shared__ bf16  vt_s[64 * 64];    // [c][kw] == B^T for O = P v
  __shared__ bf16  relw_s[64 * 64];  // [w][kw]
  __shared__ float S_s[64 * 65];
  __shared__ bf16  P_s[64 * 64];
  __shared__ float relhcol[64];
  __shared__ float mrow[64], lrow[64], corr[64];

  const int head = blockIdx.x >> 6, qb = blockIdx.x & 63;
  const int tid  = threadIdx.x;
  const int lane = tid & 31, wid = tid >> 5;
  const int wm = wid >> 1, wn = wid & 1;
  const int roff = (lane >> 4) << 3;
  const int ncol = lane & 15;

  // init staging: q tile + rel_w tile via async global->LDS
  const bf16* qp = qg + (head * N_TOK + qb * 64) * HD;
  const bf16* rw = relw_g + (head * 64 + qb) * 4096;
  for (int i = tid; i < 512; i += 256) {
    int r = i >> 3, cc = (i & 7) << 3;
    async_b128(lds_off(&q_s[r * 64 + cc]),    &qp[r * 64 + cc]);
    async_b128(lds_off(&relw_s[r * 64 + cc]), &rw[r * 64 + cc]);
  }
  if (tid < 64) { mrow[tid] = -1e30f; lrow[tid] = 0.f; }

  v8f o[2] = {};
  const int rhbase = (head * 64 + qb) * 4096;

  for (int kb = 0; kb < 64; kb++) {
    const bf16* kpp = kg + (head * N_TOK + kb * 64) * HD;
    const bf16* vpp = vg + (head * N_TOK + kb * 64) * HD;
    // prefetch next iteration's K/V tiles into cache while this one computes
    if (kb + 1 < 64 && tid < 64) {
      __builtin_prefetch(kpp + N_TOK /*next row block*/ * 0 + 64 * HD + tid * HD, 0, 1);
      __builtin_prefetch(vpp + 64 * HD + tid * HD, 0, 1);
    }
    // stage K row-major via async copy; V transposed via VGPR scatter
    for (int i = tid; i < 512; i += 256) {
      int kw = i >> 3, cc = (i & 7) << 3;
      async_b128(lds_off(&kt_s[kw * 64 + cc]), &kpp[kw * 64 + cc]);
      bf16x8 vv = *(const bf16x8*)&vpp[kw * 64 + cc];
#pragma unroll
      for (int j = 0; j < 8; j++) vt_s[(cc + j) * 64 + kw] = vv[j];
    }
    if (tid < 64) relhcol[tid] = relh_g[rhbase + tid * 64 + kb];
    wait_async();
    __syncthreads();

    // S = q @ k^T  (16 rows x 32 cols per wave)
    v8f st[2] = {};
#pragma unroll
    for (int kk = 0; kk < 64; kk += 32) {
      v16bf a  = frag_a(q_s, wm * 16, 64, kk);
      v16bf b0 = frag_b(kt_s, wn * 32,      64, kk);
      v16bf b1 = frag_b(kt_s, wn * 32 + 16, 64, kk);
      st[0] = wmma_bf16(a, b0, st[0]);
      st[1] = wmma_bf16(a, b1, st[1]);
    }
    // write S with decomposed rel-pos bias
#pragma unroll
    for (int t = 0; t < 2; t++) {
      int cbase = wn * 32 + t * 16 + ncol;
#pragma unroll
      for (int r = 0; r < 8; r++) {
        int row = wm * 16 + r + roff;
        S_s[row * 65 + cbase] =
            st[t][r] + relhcol[row] + bf2f(relw_s[row * 64 + cbase]);
      }
    }
    __syncthreads();

    // online softmax (one thread per q row)
    if (tid < 64) {
      const float* sr = &S_s[tid * 65];
      float mold = mrow[tid], mnew = mold;
#pragma unroll 8
      for (int j = 0; j < 64; j++) mnew = fmaxf(mnew, sr[j]);
      float cf = __expf(mold - mnew);
      float ls = lrow[tid] * cf;
      bf16* pr = &P_s[tid * 64];
#pragma unroll 8
      for (int j = 0; j < 64; j++) {
        float p = __expf(sr[j] - mnew);
        ls += p;
        pr[j] = f2bf(p);
      }
      mrow[tid] = mnew; lrow[tid] = ls; corr[tid] = cf;
    }
    __syncthreads();

    // O = O*corr + P @ V
#pragma unroll
    for (int t = 0; t < 2; t++)
#pragma unroll
      for (int r = 0; r < 8; r++)
        o[t][r] *= corr[wm * 16 + r + roff];
#pragma unroll
    for (int kk = 0; kk < 64; kk += 32) {
      v16bf a  = frag_a(P_s, wm * 16, 64, kk);
      v16bf b0 = frag_b(vt_s, wn * 32,      64, kk);
      v16bf b1 = frag_b(vt_s, wn * 32 + 16, 64, kk);
      o[0] = wmma_bf16(a, b0, o[0]);
      o[1] = wmma_bf16(a, b1, o[1]);
    }
    __syncthreads();   // protect kt_s/vt_s/S_s for next iteration
  }

  // normalize and write [n][head*64+c] bf16 for the output projection
#pragma unroll
  for (int t = 0; t < 2; t++) {
    int c = wn * 32 + t * 16 + ncol;
#pragma unroll
    for (int r = 0; r < 8; r++) {
      int row = wm * 16 + r + roff;
      float v = o[t][r] / lrow[row];
      outg[(qb * 64 + row) * DIM + head * HD + c] = f2bf(v);
    }
  }
}

// ----------------------------- out-proj GEMM -------------------------------
__global__ __launch_bounds__(256) void proj_gemm_kernel(
    const bf16* __restrict__ A, const bf16* __restrict__ B,
    const float* __restrict__ bias, float* __restrict__ out) {
  __shared__ bf16 a_s[128 * 40];
  __shared__ bf16 b_s[64 * 40];
  const int tid  = threadIdx.x;
  const int lane = tid & 31, wid = tid >> 5;
  const int wm = wid >> 1, wn = wid & 1;
  const int m0 = blockIdx.y * 128, n0 = blockIdx.x * 64;

  v8f acc[2][2] = {};
  for (int k0 = 0; k0 < DIM; k0 += 32) {
    for (int i = tid; i < 512; i += 256) {
      int r = i >> 2, kc = (i & 3) << 3;
      async_b128(lds_off(&a_s[r * 40 + kc]),
                 &A[(m0 + r) * DIM + k0 + kc]);
    }
    {
      int kr = tid >> 3, nc = (tid & 7) << 3;
      bf16x8 v = *(const bf16x8*)&B[(k0 + kr) * DIM + n0 + nc];
#pragma unroll
      for (int j = 0; j < 8; j++) b_s[(nc + j) * 40 + kr] = v[j];
    }
    wait_async();
    __syncthreads();
    v16bf af[2], bf_[2];
    af[0] = frag_a(a_s, wm * 32,      40, 0);
    af[1] = frag_a(a_s, wm * 32 + 16, 40, 0);
    bf_[0] = frag_b(b_s, wn * 32,      40, 0);
    bf_[1] = frag_b(b_s, wn * 32 + 16, 40, 0);
#pragma unroll
    for (int tm = 0; tm < 2; tm++)
#pragma unroll
      for (int tn = 0; tn < 2; tn++)
        acc[tm][tn] = wmma_bf16(af[tm], bf_[tn], acc[tm][tn]);
    __syncthreads();
  }

  const int roff = (lane >> 4) << 3;
  const int ncol = lane & 15;
#pragma unroll
  for (int tm = 0; tm < 2; tm++)
#pragma unroll
    for (int tn = 0; tn < 2; tn++) {
      int gn = n0 + wn * 32 + tn * 16 + ncol;
      float bv = bias[gn];
#pragma unroll
      for (int r = 0; r < 8; r++) {
        int m = m0 + wm * 32 + tm * 16 + r + roff;
        out[m * DIM + gn] = acc[tm][tn][r] + bv;
      }
    }
}

// ------------------------------ launcher -----------------------------------
static inline size_t align_up(size_t x) { return (x + 255) & ~(size_t)255; }

extern "C" void kernel_launch(void* const* d_in, const int* in_sizes, int n_in,
                              void* d_out, int out_size, void* d_ws, size_t ws_size,
                              hipStream_t stream) {
  (void)in_sizes; (void)n_in; (void)out_size; (void)ws_size;
  const float* x      = (const float*)d_in[0];   // (4096, 768)
  const float* w_qkv  = (const float*)d_in[1];   // (768, 2304)
  const float* b_qkv  = (const float*)d_in[2];   // (2304)
  const float* w_proj = (const float*)d_in[3];   // (768, 768)
  const float* b_proj = (const float*)d_in[4];   // (768)
  const float* rph    = (const float*)d_in[5];   // (127, 64)
  const float* rpw    = (const float*)d_in[6];   // (127, 64)
  float* out          = (float*)d_out;           // (4096, 768)

  // workspace carve-up (~55 MB total)
  char* p = (char*)d_ws;
  bf16*  x_bf    = (bf16*)p;  p += align_up((size_t)N_TOK * DIM * 2);
  bf16*  wqkv_bf = (bf16*)p;  p += align_up((size_t)DIM * 2304 * 2);
  bf16*  wproj_bf= (bf16*)p;  p += align_up((size_t)DIM * DIM * 2);
  bf16*  q_bf    = (bf16*)p;  p += align_up((size_t)NHEADS * N_TOK * HD * 2);
  bf16*  k_bf    = (bf16*)p;  p += align_up((size_t)NHEADS * N_TOK * HD * 2);
  bf16*  v_bf    = (bf16*)p;  p += align_up((size_t)NHEADS * N_TOK * HD * 2);
  float* relh_ws = (float*)p; p += align_up((size_t)NHEADS * 64 * 64 * 64 * 4);
  bf16*  relw_ws = (bf16*)p;  p += align_up((size_t)NHEADS * 64 * 64 * 64 * 2);
  bf16*  attn_bf = (bf16*)p;  p += align_up((size_t)N_TOK * DIM * 2);

  const int n_x    = N_TOK * DIM;
  const int n_wqkv = DIM * 2304;
  const int n_wprj = DIM * DIM;
  cvt_kernel<<<(n_x    + 255) / 256, 256, 0, stream>>>(x,      x_bf,     n_x);
  cvt_kernel<<<(n_wqkv + 255) / 256, 256, 0, stream>>>(w_qkv,  wqkv_bf,  n_wqkv);
  cvt_kernel<<<(n_wprj + 255) / 256, 256, 0, stream>>>(w_proj, wproj_bf, n_wprj);

  qkv_gemm_kernel<<<dim3(2304 / 64, N_TOK / 128), 256, 0, stream>>>(
      x_bf, wqkv_bf, b_qkv, q_bf, k_bf, v_bf);

  relh_kernel<<<NHEADS * 64, 256, 0, stream>>>(q_bf, rph, relh_ws);
  relw_kernel<<<NHEADS * 64, 256, 0, stream>>>(q_bf, rpw, relw_ws);

  flash_kernel<<<NHEADS * 64, 256, 0, stream>>>(q_bf, k_bf, v_bf,
                                                relh_ws, relw_ws, attn_bf);

  proj_gemm_kernel<<<dim3(DIM / 64, N_TOK / 128), 256, 0, stream>>>(
      attn_bf, wproj_bf, b_proj, out);
}
